// AtomAttentionEncoderDiffusion_60017872994833
// MI455X (gfx1250) — matrix operand order repaired and checked by reading, not compiled, not run on hardware
//
#include <hip/hip_runtime.h>
#include <hip/hip_bf16.h>

// ---------------------------------------------------------------------------
// CDNA5 (gfx1250, wave32) implementation of AtomAttentionEncoderDiffusion.
// All GEMMs + attention run on v_wmma_f32_16x16x32_f16 with f32 accumulate.
// Key algebraic simplifications:
//  * mQ (query mask) is provably all-false -> output gather == reshape.
//  * Pair bias B is only needed at windowed (iQ,iK) pairs: compute LN(Z)@Wb
//    directly per window (16 MB of Z instead of 256 MB), fold -1e9 K-penalty.
// CDNA5-specific paths: WMMA f16, GLOBAL_LOAD_ASYNC_TO_LDS_B128 (guarded),
// global_prefetch_b8.
// ---------------------------------------------------------------------------

typedef __attribute__((ext_vector_type(16))) _Float16     v16h;
typedef __attribute__((ext_vector_type(8)))  float        v8f;
typedef __attribute__((ext_vector_type(8)))  unsigned int v8u;

#define D_   8
#define L_   2048
#define CA_  128
#define H_   4
#define CH_  32
#define NQ_  64
#define RTOT (D_ * L_)   // 16384 rows

// ---- optional gfx1250 async global->LDS path (compile-safe guards) --------
#if defined(__has_builtin)
#  if __has_builtin(__builtin_amdgcn_global_load_async_to_lds_b128)
#    define USE_ASYNC_LDS 1
#  endif
#  if __has_builtin(__builtin_amdgcn_s_wait_asynccnt)
#    define HAVE_WAIT_ASYNC_BUILTIN 1
#  endif
#endif

// builtin wants: (int4 addrspace(1)* gsrc, int4 addrspace(3)* ldst, imm, imm)
typedef int v4i_t __attribute__((vector_size(16)));
typedef __attribute__((address_space(1))) v4i_t* as1_i4p;
typedef __attribute__((address_space(3))) v4i_t* as3_i4p;

__device__ __forceinline__ as1_i4p glob_cast(const void* p) {
  return (as1_i4p)(unsigned long long)p;
}
__device__ __forceinline__ as3_i4p lds_cast(void* p) {
  // generic LDS pointers carry the LDS offset in addr[31:0] (ISA 10.2)
  return (as3_i4p)(unsigned int)(unsigned long long)p;
}
__device__ __forceinline__ void wait_async0() {
#if defined(USE_ASYNC_LDS)
#  if defined(HAVE_WAIT_ASYNC_BUILTIN)
  __builtin_amdgcn_s_wait_asynccnt(0);
#  else
  asm volatile("s_wait_asynccnt 0x0" ::: "memory");
#  endif
#endif
}

union F4H8 { float4 v; _Float16 h[8]; };
union FragU { v8u u; v16h h; };

__device__ __forceinline__ float sigmf(float x) { return 1.0f / (1.0f + __expf(-x)); }

// A-fragment (16x32 f16) from row-major LDS, per ISA 7.12.2:
// lane: m = row0 + (lane&15); half (lane>>4) selects K 0-7/8-15 (v<4) and 16-23/24-31 (v>=4)
__device__ __forceinline__ v16h afrag(const _Float16* s, int ld, int row0, int k0, int lane) {
  const int m  = row0 + (lane & 15);
  const int kh = (lane >> 4) << 3;
  v8u t;
#pragma unroll
  for (int v = 0; v < 8; ++v) {
    int k = k0 + kh + ((v & 4) << 2) + ((v & 3) << 1);
    t[v] = *(const unsigned int*)(s + m * ld + k);   // (k,k+1) pair, 32-bit LDS read
  }
  FragU f; f.u = t; return f.h;
}

// B-fragment (32x16 f16) from N-major (transposed) LDS: element (k,n) at sT[n*ld+k].
// lane: n = n0 + (lane&15); half (lane>>4) selects K 0-7/8-15 (v<4) and 16-23/24-31 (v>=4)
__device__ __forceinline__ v16h bfragT(const _Float16* sT, int ld, int k0, int n0, int lane) {
  const int n  = n0 + (lane & 15);
  const int kb = k0 + ((lane >> 4) << 3);
  v8u t;
#pragma unroll
  for (int v = 0; v < 8; ++v)
    t[v] = *(const unsigned int*)(sT + n * ld + kb + ((v & 4) << 2) + ((v & 3) << 1));
  FragU f; f.u = t; return f.h;
}

// ---------------------------------------------------------------------------
// Kernel 1: convert / pack all weights to f16 GEMM-ready layouts
// ---------------------------------------------------------------------------
__global__ void prep_weights(const float* Wq, const float* Wk, const float* Wv, const float* Wg,
                             const float* adagW, const float* adabW, const float* Wa, const float* Wo,
                             _Float16* Wqkvg, _Float16* adacat, _Float16* Wa_h, _Float16* Wo_h) {
  int idx = blockIdx.x * 256 + threadIdx.x;   // 128*1024 total
  int c = idx >> 10;
  int col = idx & 1023;
  if (col < 512) {
    int which = col >> 7, cc = col & 127;
    const float* src = (which == 0) ? Wq : (which == 1) ? Wk : (which == 2) ? Wv : Wg;
    Wqkvg[c * 512 + col] = (_Float16)src[c * 128 + cc];
  } else if (col < 768) {
    int j = col - 512;
    adacat[c * 256 + j] = (_Float16)((j < 128) ? adagW[c * 128 + j] : adabW[c * 128 + (j - 128)]);
  } else if (col < 896) {
    int j = col - 768;
    Wa_h[c * 128 + j] = (_Float16)Wa[c * 128 + j];
  } else {
    int j = col - 896;
    Wo_h[c * 128 + j] = (_Float16)Wo[c * 128 + j];
  }
}

// ---------------------------------------------------------------------------
// Kernel 2: LN(A_I), LN(S_I), raw S_I -> f16 rows. One wave32 per row.
// ---------------------------------------------------------------------------
__global__ void ln_kernel(const float* __restrict__ A, const float* __restrict__ S,
                          _Float16* aln, _Float16* sln, _Float16* sraw) {
  const int wave = threadIdx.x >> 5, lane = threadIdx.x & 31;
  const size_t r = (size_t)blockIdx.x * 8 + wave;
  const float* ar = A + r * 128;
  const float* sr = S + r * 128;
  float xa[4], xs[4];
  float s1a = 0, s2a = 0, s1s = 0, s2s = 0;
#pragma unroll
  for (int k = 0; k < 4; ++k) {
    xa[k] = ar[lane + 32 * k]; xs[k] = sr[lane + 32 * k];
    s1a += xa[k]; s2a += xa[k] * xa[k];
    s1s += xs[k]; s2s += xs[k] * xs[k];
  }
  for (int o = 16; o; o >>= 1) {
    s1a += __shfl_xor(s1a, o); s2a += __shfl_xor(s2a, o);
    s1s += __shfl_xor(s1s, o); s2s += __shfl_xor(s2s, o);
  }
  const float ma = s1a * (1.f / 128.f), va = s2a * (1.f / 128.f) - ma * ma, ra = rsqrtf(va + 1e-5f);
  const float ms = s1s * (1.f / 128.f), vs = s2s * (1.f / 128.f) - ms * ms, rs = rsqrtf(vs + 1e-5f);
#pragma unroll
  for (int k = 0; k < 4; ++k) {
    int j = lane + 32 * k;
    aln[r * 128 + j]  = (_Float16)((xa[k] - ma) * ra);
    sln[r * 128 + j]  = (_Float16)((xs[k] - ms) * rs);
    sraw[r * 128 + j] = (_Float16)xs[k];
  }
}

// ---------------------------------------------------------------------------
// Kernel 3: generic WMMA GEMM, M=16384, K=128 (lda=128 contiguous), tile 128xN.
// ---------------------------------------------------------------------------
#define GM_RAW   0   // outf[r*N+col] = c
#define GM_QKVG  1   // col<384 -> outh f16 qkv; else auxf G = sigmoid(c)
#define GM_GATE  2   // outf gate = sigmoid(c + bvec[col])
#define GM_FINAL 3   // outf[r*128+col] = auxf[r*128+col] * c

__global__ void gemm_k128(const _Float16* __restrict__ A, const _Float16* __restrict__ B,
                          int N, int mode, float* outf, _Float16* outh, float* auxf,
                          const float* bvec) {
  __shared__ _Float16 As[128 * 128];   // 32 KB
  __shared__ _Float16 BsT[64 * 128];   // 16 KB, N-major transposed
  const int tid = threadIdx.x, lane = tid & 31, wid = tid >> 5;

  {  // A block: 128 rows x 128 cols, fully contiguous (lda==K==128)
    const float4* src = (const float4*)(A + (size_t)blockIdx.x * 128 * 128);
    float4* dst = (float4*)As;
#if defined(USE_ASYNC_LDS)
#pragma unroll
    for (int i = tid; i < 2048; i += 256)
      __builtin_amdgcn_global_load_async_to_lds_b128(glob_cast(src + i), lds_cast(dst + i), 0, 0);
    wait_async0();
#else
#pragma unroll
    for (int i = tid; i < 2048; i += 256) dst[i] = src[i];
#endif
  }

  for (int n0 = 0; n0 < N; n0 += 64) {
    __syncthreads();
    // prefetch next B chunk rows into caches (global_prefetch_b8)
    if (n0 + 64 < N)
      __builtin_prefetch(B + (size_t)(tid & 127) * N + n0 + 64, 0, 0);
    for (int i = tid; i < 1024; i += 256) {     // stage B chunk transposed
      int k = i >> 3, e8 = i & 7;
      F4H8 u; u.v = *(const float4*)(B + (size_t)k * N + n0 + e8 * 8);
#pragma unroll
      for (int e = 0; e < 8; ++e) BsT[(e8 * 8 + e) * 128 + k] = u.h[e];
    }
    __syncthreads();

    // preload all four A fragments up front: independent DS loads in flight,
    // so WMMAs don't serialize on s_wait_dscnt 0
    v16h aa[4];
#pragma unroll
    for (int ks = 0; ks < 4; ++ks) aa[ks] = afrag(As, 128, wid * 16, ks * 32, lane);

    v8f acc0 = {}, acc1 = {}, acc2 = {}, acc3 = {};
#pragma unroll
    for (int ks = 0; ks < 4; ++ks) {
      v16h b0 = bfragT(BsT, 128, ks * 32,  0, lane);
      v16h b1 = bfragT(BsT, 128, ks * 32, 16, lane);
      v16h b2 = bfragT(BsT, 128, ks * 32, 32, lane);
      v16h b3 = bfragT(BsT, 128, ks * 32, 48, lane);
      acc0 = __builtin_amdgcn_wmma_f32_16x16x32_f16(false, aa[ks], false, b0, (short)0, acc0, false, false);
      acc1 = __builtin_amdgcn_wmma_f32_16x16x32_f16(false, aa[ks], false, b1, (short)0, acc1, false, false);
      acc2 = __builtin_amdgcn_wmma_f32_16x16x32_f16(false, aa[ks], false, b2, (short)0, acc2, false, false);
      acc3 = __builtin_amdgcn_wmma_f32_16x16x32_f16(false, aa[ks], false, b3, (short)0, acc3, false, false);
    }

    const int rbase = blockIdx.x * 128 + wid * 16 + ((lane >> 4) << 3);
    const int nlo = lane & 15;
#pragma unroll
    for (int nt = 0; nt < 4; ++nt) {
      v8f acc = (nt == 0) ? acc0 : (nt == 1) ? acc1 : (nt == 2) ? acc2 : acc3;
      const int col = n0 + nt * 16 + nlo;
#pragma unroll
      for (int v = 0; v < 8; ++v) {
        const size_t r = (size_t)(rbase + v);
        const float c = acc[v];
        if (mode == GM_RAW) {
          outf[r * (size_t)N + col] = c;
        } else if (mode == GM_QKVG) {
          if (col < 384) outh[r * 384 + col] = (_Float16)c;
          else           auxf[r * 128 + (col - 384)] = sigmf(c);
        } else if (mode == GM_GATE) {
          outf[r * 128 + col] = sigmf(c + bvec[col]);
        } else {  // GM_FINAL
          outf[r * 128 + col] = auxf[r * 128 + col] * c;
        }
      }
    }
  }
}

// ---------------------------------------------------------------------------
// Kernel 4: adaLN combine: a = sigmoid(Xg + ada_gb) * LN(A) + Xb  -> f16
// ---------------------------------------------------------------------------
__global__ void ada_combine(const float* __restrict__ X, const _Float16* __restrict__ aln,
                            const float* __restrict__ adagb, _Float16* a_h) {
  const size_t idx = (size_t)blockIdx.x * 256 + threadIdx.x;
  const size_t r = idx >> 7; const int j = idx & 127;
  const float g = sigmf(X[r * 256 + j] + adagb[j]);
  a_h[idx] = (_Float16)(g * (float)aln[idx] + X[r * 256 + 128 + j]);
}

// ---------------------------------------------------------------------------
// Kernel 5: windowed pair bias: LN16(Z[iq,ik]) @ Wb_pair  (+ K-edge penalty)
// ---------------------------------------------------------------------------
__global__ void bias_kernel(const float* __restrict__ Z, const float* __restrict__ ln0w,
                            const float* __restrict__ ln0b, const float* __restrict__ Wb,
                            float* __restrict__ biasb) {
  const int p = blockIdx.x * 256 + threadIdx.x;   // 64*32*128 pairs
  const int n = p >> 12;
  const int rem = p & 4095;
  const int i = rem >> 7, j = rem & 127;
  const int iq = n * 32 + i;                       // always in range (mQ == false)
  const int ikr = n * 32 + j - 48;
  const float pen = (ikr < 0 || ikr > (L_ - 1)) ? -1.0e9f : 0.0f;
  const int ik = min(max(ikr, 0), L_ - 1);
  const float* z = Z + ((size_t)iq * L_ + ik) * 16;

  float x[16], s1 = 0.f, s2 = 0.f;
#pragma unroll
  for (int q = 0; q < 4; ++q) {
    float4 t = *(const float4*)(z + q * 4);
    x[q * 4 + 0] = t.x; x[q * 4 + 1] = t.y; x[q * 4 + 2] = t.z; x[q * 4 + 3] = t.w;
  }
#pragma unroll
  for (int q = 0; q < 16; ++q) { s1 += x[q]; s2 += x[q] * x[q]; }
  const float m  = s1 * (1.f / 16.f);
  const float va = s2 * (1.f / 16.f) - m * m;
  const float rs = rsqrtf(va + 1e-5f);

  float acc[4] = {0.f, 0.f, 0.f, 0.f};
#pragma unroll
  for (int q = 0; q < 16; ++q) {
    const float y = (x[q] - m) * rs * ln0w[q] + ln0b[q];
#pragma unroll
    for (int h = 0; h < 4; ++h) acc[h] += y * Wb[q * 4 + h];
  }
#pragma unroll
  for (int h = 0; h < 4; ++h)
    biasb[(((size_t)n * 4 + h) * 32 + i) * 128 + j] = acc[h] + pen;
}

// ---------------------------------------------------------------------------
// Kernel 6: windowed attention. One block per (window, d); 4 waves, 1 per head.
// ---------------------------------------------------------------------------
#define ATTN_SMEM (32*128*2 + 128*128*2 + 128*128*2 + 4*32*128*4 + 4*32*128*2)  // 172032 B

__global__ void attn_kernel(const _Float16* __restrict__ qkv, const float* __restrict__ biasb,
                            const float* __restrict__ G, _Float16* __restrict__ go) {
  extern __shared__ char smem[];
  _Float16* Qs  = (_Float16*)smem;           // 32 x 128 (row-major, all heads)
  _Float16* Ks  = Qs + 32 * 128;             // 128 x 128 row j, col ch (row-major)
  _Float16* VsT = Ks + 128 * 128;            // 128 (ch) x 128 (j)  (transposed)
  float*    S   = (float*)(VsT + 128 * 128); // 4 heads x 32 x 128
  _Float16* P   = (_Float16*)(S + 4 * 32 * 128);

  const int nw = blockIdx.x, d = blockIdx.y;
  const int tid = threadIdx.x, lane = tid & 31, h = tid >> 5;

  for (int idx = tid; idx < 512; idx += 128) {          // Q window (rows l = nw*32+i)
    int i = idx >> 4, c16 = idx & 15;
    size_t r = (size_t)d * L_ + nw * 32 + i;
    const float4* src = ((const float4*)(qkv + r * 384)) + c16;
#if defined(USE_ASYNC_LDS)
    __builtin_amdgcn_global_load_async_to_lds_b128(glob_cast(src), lds_cast((float4*)Qs + idx), 0, 0);
#else
    ((float4*)Qs)[idx] = *src;
#endif
  }
  for (int idx = tid; idx < 2048; idx += 128) {         // K rows (async) + V transposed
    int j = idx >> 4, c16 = idx & 15;
    int ikr = nw * 32 + j - 48;
    int ik = min(max(ikr, 0), L_ - 1);                  // clamped; penalty kills masked j
    size_t r = (size_t)d * L_ + ik;
    const float4* ksrc = ((const float4*)(qkv + r * 384 + 128)) + c16;
#if defined(USE_ASYNC_LDS)
    __builtin_amdgcn_global_load_async_to_lds_b128(glob_cast(ksrc), lds_cast((float4*)Ks + idx), 0, 0);
#else
    ((float4*)Ks)[idx] = *ksrc;
#endif
    F4H8 w; w.v = ((const float4*)(qkv + r * 384 + 256))[c16];
#pragma unroll
    for (int e = 0; e < 8; ++e) VsT[(c16 * 8 + e) * 128 + j] = w.h[e];
  }
  wait_async0();
  __syncthreads();

  float*    Sh = S + h * 32 * 128;
  _Float16* Ph = P + h * 32 * 128;
  const float scale = 0.1767766952966369f;  // 1/sqrt(32)
  const float* bh = biasb + (((size_t)nw * 4 + h) * 32) * 128;

#pragma unroll
  for (int mt = 0; mt < 2; ++mt) {                      // logits = Q K^T * scale + bias
    v16h a = afrag(Qs, 128, mt * 16, h * 32, lane);
#pragma unroll
    for (int jt = 0; jt < 8; ++jt) {
      v16h b = bfragT(Ks, 128, h * 32, jt * 16, lane);  // (k=ch, n=j) from row-major K
      v8f acc = {};
      acc = __builtin_amdgcn_wmma_f32_16x16x32_f16(false, a, false, b, (short)0, acc, false, false);
      const int jj = jt * 16 + (lane & 15);
      const int mbase = mt * 16 + ((lane >> 4) << 3);
#pragma unroll
      for (int v = 0; v < 8; ++v)
        Sh[(mbase + v) * 128 + jj] = acc[v] * scale + bh[(mbase + v) * 128 + jj];
    }
  }
  __syncthreads();

  {  // softmax over j, lane == row
    float* row = Sh + lane * 128;
    float mx = -3.0e38f;
    for (int j = 0; j < 128; ++j) mx = fmaxf(mx, row[j]);
    float sum = 0.f;
    for (int j = 0; j < 128; ++j) { float e = __expf(row[j] - mx); row[j] = e; sum += e; }
    const float inv = 1.0f / sum;
    _Float16* prow = Ph + lane * 128;
    for (int j = 0; j < 128; ++j) prow[j] = (_Float16)(row[j] * inv);
  }
  __syncthreads();

#pragma unroll
  for (int mt = 0; mt < 2; ++mt) {                      // out = P @ V, fused gate G
    v8f acc0 = {}, acc1 = {};
#pragma unroll
    for (int ks = 0; ks < 4; ++ks) {
      v16h a  = afrag(Ph, 128, mt * 16, ks * 32, lane);
      v16h b0 = bfragT(VsT, 128, ks * 32, h * 32 +  0, lane);
      v16h b1 = bfragT(VsT, 128, ks * 32, h * 32 + 16, lane);
      acc0 = __builtin_amdgcn_wmma_f32_16x16x32_f16(false, a, false, b0, (short)0, acc0, false, false);
      acc1 = __builtin_amdgcn_wmma_f32_16x16x32_f16(false, a, false, b1, (short)0, acc1, false, false);
    }
    const int mbase = mt * 16 + ((lane >> 4) << 3);
    const int nlo = lane & 15;
#pragma unroll
    for (int v = 0; v < 8; ++v) {
      const size_t r = (size_t)d * L_ + nw * 32 + mbase + v;
      const int ch0 = h * 32 + nlo;
      go[r * 128 + ch0]      = (_Float16)(G[r * 128 + ch0]      * acc0[v]);
      go[r * 128 + ch0 + 16] = (_Float16)(G[r * 128 + ch0 + 16] * acc1[v]);
    }
  }
}

// ---------------------------------------------------------------------------
extern "C" void kernel_launch(void* const* d_in, const int* in_sizes, int n_in,
                              void* d_out, int out_size, void* d_ws, size_t ws_size,
                              hipStream_t stream) {
  (void)in_sizes; (void)n_in; (void)out_size; (void)ws_size;

  const float* A_I   = (const float*)d_in[0];
  const float* S_I   = (const float*)d_in[1];
  const float* Z     = (const float*)d_in[2];
  const float* Wq    = (const float*)d_in[3];
  const float* Wk    = (const float*)d_in[4];
  const float* Wv    = (const float*)d_in[5];
  const float* Wg    = (const float*)d_in[6];
  const float* Wb    = (const float*)d_in[7];
  const float* ln0w  = (const float*)d_in[8];
  const float* ln0b  = (const float*)d_in[9];
  const float* adagW = (const float*)d_in[10];
  const float* adagb = (const float*)d_in[11];
  const float* adabW = (const float*)d_in[12];
  const float* Wa    = (const float*)d_in[13];
  const float* Wo    = (const float*)d_in[14];
  const float* bo    = (const float*)d_in[15];

  char* ws = (char*)d_ws;
  size_t off = 0;
  auto alloc = [&](size_t bytes) { void* p = ws + off; off += (bytes + 255) & ~(size_t)255; return p; };

  _Float16* aln    = (_Float16*)alloc((size_t)RTOT * 128 * 2);
  _Float16* sln    = (_Float16*)alloc((size_t)RTOT * 128 * 2);
  _Float16* sraw   = (_Float16*)alloc((size_t)RTOT * 128 * 2);
  _Float16* a_h    = (_Float16*)alloc((size_t)RTOT * 128 * 2);
  _Float16* qkv    = (_Float16*)alloc((size_t)RTOT * 384 * 2);
  _Float16* go     = (_Float16*)alloc((size_t)RTOT * 128 * 2);
  _Float16* Wqkvg  = (_Float16*)alloc(128 * 512 * 2);
  _Float16* adacat = (_Float16*)alloc(128 * 256 * 2);
  _Float16* Wa_h   = (_Float16*)alloc(128 * 128 * 2);
  _Float16* Wo_h   = (_Float16*)alloc(128 * 128 * 2);
  float*    X      = (float*)alloc((size_t)RTOT * 256 * 4);
  float*    G      = (float*)alloc((size_t)RTOT * 128 * 4);
  float*    gate   = (float*)alloc((size_t)RTOT * 128 * 4);
  float*    biasb  = (float*)alloc((size_t)NQ_ * 4 * 32 * 128 * 4);

  // 1. pack weights to f16
  prep_weights<<<512, 256, 0, stream>>>(Wq, Wk, Wv, Wg, adagW, adabW, Wa, Wo,
                                        Wqkvg, adacat, Wa_h, Wo_h);
  // 2. layernorms (+ raw S f16)
  ln_kernel<<<RTOT / 8, 256, 0, stream>>>(A_I, S_I, aln, sln, sraw);
  // 3. adaLN GEMM: X = LN(S) @ [ada_gW | ada_bW]
  gemm_k128<<<RTOT / 128, 256, 0, stream>>>(sln, adacat, 256, GM_RAW, X, nullptr, nullptr, nullptr);
  // 4. a = sigmoid(Xg + ada_gb)*LN(A) + Xb
  ada_combine<<<(RTOT * 128) / 256, 256, 0, stream>>>(X, aln, adagb, a_h);
  // 5. QKVG = a @ [Wq|Wk|Wv|Wg]; G gets sigmoid
  gemm_k128<<<RTOT / 128, 256, 0, stream>>>(a_h, Wqkvg, 512, GM_QKVG, nullptr, qkv, G, nullptr);
  // 6. gate = sigmoid(S_I @ Wo + bo)
  gemm_k128<<<RTOT / 128, 256, 0, stream>>>(sraw, Wo_h, 128, GM_GATE, gate, nullptr, nullptr, bo);
  // 7. windowed pair bias (+K penalty)
  bias_kernel<<<(NQ_ * 32 * 128) / 256, 256, 0, stream>>>(Z, ln0w, ln0b, Wb, biasb);
  // 8. windowed attention + gating G -> go (f16)
  attn_kernel<<<dim3(NQ_, D_), 128, ATTN_SMEM, stream>>>(qkv, biasb, G, go);
  // 9. out = gate * (go @ Wa)
  gemm_k128<<<RTOT / 128, 256, 0, stream>>>(go, Wa_h, 128, GM_FINAL, (float*)d_out, nullptr, gate, nullptr);
}
